// groupedLinearDecoder_4097398801011
// MI455X (gfx1250) — matrix operand order repaired and checked by reading, not compiled, not run on hardware
//
#include <hip/hip_runtime.h>
#include <hip/hip_bf16.h>

// ---------------------------------------------------------------------------
// Grouped linear decoder: out[b,l,:] = code[b,l,:] @ W[l,:,:]
//   code: [B=256, L=64, C=256] f32
//   W:    [L=64, C=256, HW=1024] f32
//   out:  [B=256, L=64, HW=1024] f32
//
// Roofline: 8.6 GFLOP vs 151 MB -> ~6.5 us HBM floor @ 23.3 TB/s; fp32 WMMA
// (K=4/inst) is ~2.7x more instructions than split-bf16 emulation, so use
// 3x V_WMMA_F32_16X16X32_BF16 (hi*hi + hi*lo + lo*hi) per K=32 slab, which
// keeps near-fp32 accuracy at bf16-class matrix throughput.
//
// Software pipeline: global loads for slab k+1 are issued right after the
// staging barrier of slab k, hiding HBM latency behind the 24-WMMA compute.
// ---------------------------------------------------------------------------

typedef __attribute__((ext_vector_type(16))) __bf16 v16bf;
typedef __attribute__((ext_vector_type(8)))  __bf16 v8bf;
typedef __attribute__((ext_vector_type(8)))  float  v8f;

#define B_DIM  256
#define L_DIM  64
#define C_DIM  256
#define HW_DIM 1024
#define MT 128
#define NT 128
#define KT 32

union FragBF {
    v16bf v;
    v8bf  h[2];
};

__device__ __forceinline__ unsigned short f32_to_bf16_rne(float x) {
    unsigned u = __float_as_uint(x);
    unsigned r = u + 0x7FFFu + ((u >> 16) & 1u);
    return (unsigned short)(r >> 16);
}
__device__ __forceinline__ float bf16_bits_to_f32(unsigned short h) {
    return __uint_as_float(((unsigned)h) << 16);
}

__global__ __launch_bounds__(256) void grouped_linear_wmma(
    const float* __restrict__ code,
    const float* __restrict__ W,
    float* __restrict__ out)
{
    __shared__ unsigned short Ah[MT][KT];   // code tile, bf16 high part, [m][k]
    __shared__ unsigned short Al[MT][KT];   // code tile, bf16 low part
    __shared__ unsigned short Bh[NT][KT];   // W tile transposed, [n][k]
    __shared__ unsigned short Bl[NT][KT];

    const int l   = blockIdx.z;
    const int m0  = blockIdx.y * MT;
    const int n0  = blockIdx.x * NT;
    const int tid = threadIdx.x;
    const int lane   = tid & 31;
    const int lane15 = lane & 15;
    const int wid = tid >> 5;
    const int wm  = wid >> 1;   // 0..3 : 32-row band
    const int wn  = wid & 1;    // 0..1 : 64-col band

    // A fragment K offsets per ISA 7.12.2 (16-bit A 16x32 layout)
    const int ka = (lane < 16) ? 0 : 8;    // halves at [ka..ka+7] and [ka+16..ka+23]
    // B fragment: lane holds one column, K halves at [kb..kb+7] and [kb+16..kb+23]
    const int kb = (lane < 16) ? 0 : 8;

    // ---- fixed per-thread staging coordinates ----
    int arow[4], akc[4], brow[4], bnc[4];
    const float* abase[4];
    const float* bbase[4];
    #pragma unroll
    for (int i = 0; i < 4; ++i) {
        int id  = i * 256 + tid;            // 1024 float4 chunks per tile
        arow[i] = id >> 3;                  // A: 8 float4 per row (KT=32 floats)
        akc[i]  = (id & 7) << 2;
        brow[i] = id >> 5;                  // B: 32 float4 per k-row (NT=128 floats)
        bnc[i]  = (id & 31) << 2;
        abase[i] = code + ((size_t)(m0 + arow[i]) * L_DIM + l) * C_DIM + akc[i];
        bbase[i] = W + ((size_t)l * C_DIM + brow[i]) * HW_DIM + n0 + bnc[i];
    }

    v8f acc[2][4];
    #pragma unroll
    for (int mi = 0; mi < 2; ++mi)
        #pragma unroll
        for (int ni = 0; ni < 4; ++ni)
            #pragma unroll
            for (int e = 0; e < 8; ++e)
                acc[mi][ni][e] = 0.0f;

    // ---- pipeline prologue: fetch slab 0 into registers ----
    float4 areg[4], breg[4];
    #pragma unroll
    for (int i = 0; i < 4; ++i) {
        areg[i] = *reinterpret_cast<const float4*>(abase[i]);
        breg[i] = *reinterpret_cast<const float4*>(bbase[i]);
    }

    for (int k0 = 0; k0 < C_DIM; k0 += KT) {
        // ---- split fp32 -> bf16 hi/lo and store to LDS ----
        #pragma unroll
        for (int i = 0; i < 4; ++i) {
            float av[4] = {areg[i].x, areg[i].y, areg[i].z, areg[i].w};
            #pragma unroll
            for (int j = 0; j < 4; ++j) {
                unsigned short h = f32_to_bf16_rne(av[j]);
                float lo = av[j] - bf16_bits_to_f32(h);
                Ah[arow[i]][akc[i] + j] = h;
                Al[arow[i]][akc[i] + j] = f32_to_bf16_rne(lo);
            }
            float bv[4] = {breg[i].x, breg[i].y, breg[i].z, breg[i].w};
            #pragma unroll
            for (int j = 0; j < 4; ++j) {
                unsigned short h = f32_to_bf16_rne(bv[j]);
                float lo = bv[j] - bf16_bits_to_f32(h);
                Bh[bnc[i] + j][brow[i]] = h;       // transposed: [n][k]
                Bl[bnc[i] + j][brow[i]] = f32_to_bf16_rne(lo);
            }
        }

        __syncthreads();

        // ---- issue next-slab global loads now; latency hides under WMMA ----
        if (k0 + KT < C_DIM) {
            const int kn = k0 + KT;
            #pragma unroll
            for (int i = 0; i < 4; ++i) {
                areg[i] = *reinterpret_cast<const float4*>(abase[i] + kn);
                breg[i] = *reinterpret_cast<const float4*>(bbase[i] + (size_t)kn * HW_DIM);
            }
        }

        // ---- load fragments from LDS ----
        FragBF ahf[2], alf[2], bhf[4], blf[4];
        #pragma unroll
        for (int mi = 0; mi < 2; ++mi) {
            int row = wm * 32 + mi * 16 + lane15;
            ahf[mi].h[0] = *reinterpret_cast<const v8bf*>(&Ah[row][ka]);
            ahf[mi].h[1] = *reinterpret_cast<const v8bf*>(&Ah[row][ka + 16]);
            alf[mi].h[0] = *reinterpret_cast<const v8bf*>(&Al[row][ka]);
            alf[mi].h[1] = *reinterpret_cast<const v8bf*>(&Al[row][ka + 16]);
        }
        #pragma unroll
        for (int ni = 0; ni < 4; ++ni) {
            int col = wn * 64 + ni * 16 + lane15;
            bhf[ni].h[0] = *reinterpret_cast<const v8bf*>(&Bh[col][kb]);
            bhf[ni].h[1] = *reinterpret_cast<const v8bf*>(&Bh[col][kb + 16]);
            blf[ni].h[0] = *reinterpret_cast<const v8bf*>(&Bl[col][kb]);
            blf[ni].h[1] = *reinterpret_cast<const v8bf*>(&Bl[col][kb + 16]);
        }

        // ---- split-bf16 emulated-fp32 WMMA: hi*hi + hi*lo + lo*hi ----
        #pragma unroll
        for (int mi = 0; mi < 2; ++mi) {
            #pragma unroll
            for (int ni = 0; ni < 4; ++ni) {
                acc[mi][ni] = __builtin_amdgcn_wmma_f32_16x16x32_bf16(
                    false, ahf[mi].v, false, bhf[ni].v, (short)0, acc[mi][ni], false, false);
                acc[mi][ni] = __builtin_amdgcn_wmma_f32_16x16x32_bf16(
                    false, ahf[mi].v, false, blf[ni].v, (short)0, acc[mi][ni], false, false);
                acc[mi][ni] = __builtin_amdgcn_wmma_f32_16x16x32_bf16(
                    false, alf[mi].v, false, bhf[ni].v, (short)0, acc[mi][ni], false, false);
            }
        }

        __syncthreads();
    }

    // ---- store: C/D layout — VGPR r holds row r + (lane<16 ? 0 : 8), col = lane&15 ----
    const int moff = (lane < 16) ? 0 : 8;
    const int mrowbase = m0 + wm * 32;
    const int nbase    = n0 + wn * 64;
    #pragma unroll
    for (int mi = 0; mi < 2; ++mi) {
        #pragma unroll
        for (int ni = 0; ni < 4; ++ni) {
            #pragma unroll
            for (int r = 0; r < 8; ++r) {
                int m = mrowbase + mi * 16 + moff + r;
                int n = nbase + ni * 16 + lane15;
                out[((size_t)m * L_DIM + l) * HW_DIM + n] = acc[mi][ni][r];
            }
        }
    }
}

extern "C" void kernel_launch(void* const* d_in, const int* in_sizes, int n_in,
                              void* d_out, int out_size, void* d_ws, size_t ws_size,
                              hipStream_t stream) {
    const float* code = (const float*)d_in[0];
    const float* W    = (const float*)d_in[1];
    float* out        = (float*)d_out;

    dim3 grid(HW_DIM / NT, B_DIM / MT, L_DIM);  // (8, 2, 64)
    dim3 block(256);
    grouped_linear_wmma<<<grid, block, 0, stream>>>(code, W, out);
}
